// CrossAttention2D_38774964748372
// MI455X (gfx1250) — matrix operand order repaired and checked by reading, not compiled, not run on hardware
//
#include <hip/hip_runtime.h>

// ---------------------------------------------------------------------------
// CrossAttention2D for MI455X (gfx1250, wave32, WMMA + TDM).
//
// Pipeline (bf16 compute, fp32 WMMA accumulators):
//   1) fp32 -> bf16 converts (x, y, 4 weights)
//   2) Q = (w_q x + b_q)/8  -> transposed [b][n][c]   (mode 1)
//      K = (w_k y + b_k)    -> row-major  [b][c][n]   (mode 0)
//      V = (w_v y + b_v)    -> transposed [b][n][c]   (mode 1)
//   3) flash attention per (b, h, 64-row q tile), online softmax
//   4) out = w_o O + b_o -> fp32 d_out                (mode 2)
//
// Row-major tiles (GEMM A-tile, attention Q-tile) are fetched with the
// Tensor Data Mover (tensor_load_to_lds + s_wait_tensorcnt) when the builtin
// exists; transposed tiles use phased wide loads + LDS scatter stores.
// ---------------------------------------------------------------------------

typedef unsigned short u16;
typedef unsigned int   u32;
typedef unsigned long long u64;
typedef __attribute__((ext_vector_type(16))) __bf16 v16bf;
typedef __attribute__((ext_vector_type(8)))  float  v8f;
typedef __attribute__((ext_vector_type(4)))  u32    v4u;
typedef __attribute__((ext_vector_type(4)))  int    v4i;
typedef __attribute__((ext_vector_type(8)))  int    v8i;

#if defined(__has_builtin)
#  if __has_builtin(__builtin_amdgcn_tensor_load_to_lds) && \
      __has_builtin(__builtin_amdgcn_s_wait_tensorcnt)
#    define USE_TDM 1
#  endif
#endif
#ifndef USE_TDM
#  define USE_TDM 0
#endif

union Frag {
  v16bf v;
  u16   u[16];
  v4u   q[2];
};

__device__ __forceinline__ u16 f2bf(float f) {
  union { float f; u32 u; } c; c.f = f;
  u32 r = c.u + 0x7FFFu + ((c.u >> 16) & 1u);   // round-to-nearest-even
  return (u16)(r >> 16);
}

// B-matrix fragment: elems 0-7 = K kb..kb+7, elems 8-15 = K kb+16..kb+23.
__device__ __forceinline__ v16bf frag_contig(const u16* p) {
  Frag f;
  f.q[0] = *(const v4u*)p;
  f.q[1] = *(const v4u*)(p + 16);
  return f.v;
}
// A-matrix fragment: elems 0-7 = K kb..kb+7, elems 8-15 = K kb+16..kb+23.
__device__ __forceinline__ v16bf frag_split(const u16* p) {
  Frag f;
  f.q[0] = *(const v4u*)p;
  f.q[1] = *(const v4u*)(p + 16);
  return f.v;
}

__device__ __forceinline__ v8f wmma_bf16(v16bf a, v16bf b, v8f c) {
  return __builtin_amdgcn_wmma_f32_16x16x32_bf16(
      false, a, false, b, (short)0, c, false, false);
}

#if USE_TDM
// 2D TDM load: tile_w elements per row (bf16), tile_h rows, row pitch in
// elements. D# packing per CDNA5 ISA section 8 (groups 0/1, 2B data size).
__device__ __forceinline__ void tdm_load_2d(u32 lds_byte_addr, const void* gptr,
                                            u32 tile_w, u32 tile_h, u32 pitch) {
  u64 ga = (u64)(size_t)gptr;
  v4u g0 = {};
  g0[0] = 1u;                                         // count=1, user D#
  g0[1] = lds_byte_addr;                              // lds_addr
  g0[2] = (u32)(ga & 0xFFFFFFFFull);                  // global_addr[31:0]
  g0[3] = (u32)((ga >> 32) & 0x1FFFFFFull) | (2u << 30);  // addr[56:32], type=2
  u32 td0 = pitch;      // tensor_dim0 (elements)
  u32 td1 = tile_h;     // tensor_dim1
  u64 s0  = pitch;      // tensor_dim0_stride (elements)
  v8i g1 = {};
  g1[0] = (int)(1u << 16);                            // data_size = 2 bytes
  g1[1] = (int)((td0 & 0xFFFFu) << 16);               // tensor_dim0[15:0]
  g1[2] = (int)((td0 >> 16) | ((td1 & 0xFFFFu) << 16));
  g1[3] = (int)((td1 >> 16) | (tile_w << 16));        // tile_dim0
  g1[4] = (int)(tile_h & 0xFFFFu);                    // tile_dim1 (tile_dim2=0)
  g1[5] = (int)(s0 & 0xFFFFFFFFull);                  // stride0[31:0]
  g1[6] = (int)((s0 >> 32) & 0xFFFFu);                // stride0[47:32]
  g1[7] = 0;
  v4i z = {};
#if __clang_major__ >= 23
  v8i z8 = {};
  __builtin_amdgcn_tensor_load_to_lds(g0, g1, z, z, z8, 0);
#else
  __builtin_amdgcn_tensor_load_to_lds(g0, g1, z, z, 0);
#endif
}
#endif

// ---------------------------------------------------------------------------
__global__ void cvt_bf16(const float* __restrict__ s, u16* __restrict__ d, int n) {
  int i = blockIdx.x * blockDim.x + threadIdx.x;
  if (i < n) d[i] = f2bf(s[i]);
}

// ---------------------------------------------------------------------------
// D = W(MxK) @ X(KxN=1024, batched) + bias.  64x64 block tile, 4 waves.
// mode 0: bf16 [b][m][n]; mode 1: bf16 [b][n][m] * scale; mode 2: fp32 [b][m][n]
// ---------------------------------------------------------------------------
__global__ __launch_bounds__(128) void gemm_wmma(
    const u16* __restrict__ W, const u16* __restrict__ X,
    const float* __restrict__ bias, void* __restrict__ out,
    int M, int K, int mode, float scale) {
  __shared__ __align__(16) u16 As[64 * 32];   // [m][k], k contiguous
  __shared__ __align__(16) u16 Bs[64 * 32];   // [n][k], k contiguous
  const int bm = blockIdx.x * 64, bn = blockIdx.y * 64, bat = blockIdx.z;
  const u16* Xb = X + (size_t)bat * K * 1024;
  const int tid = threadIdx.x, lane = tid & 31, wave = tid >> 5;
  const int l16 = lane & 15, hi = (lane >> 4) & 1;

  v8f acc[4] = {};
  for (int k0 = 0; k0 < K; k0 += 32) {
    __syncthreads();                       // previous tile fully consumed

#if USE_TDM
    if (wave == 0)                         // one TDM per workgroup, EXEC-free
      tdm_load_2d((u32)(size_t)As, W + (size_t)bm * K + k0, 32, 64, (u32)K);
#else
    {  // A tile 64x32, row-major copy: 2 x b128 per thread, phased
      v4u a_reg[2];
      for (int i = 0; i < 2; i++) {
        int idx = tid + i * 128;
        int r = idx >> 2, c8 = (idx & 3) * 8;
        a_reg[i] = *(const v4u*)(W + (size_t)(bm + r) * K + k0 + c8);
      }
      for (int i = 0; i < 2; i++) {
        int idx = tid + i * 128;
        int r = idx >> 2, c8 = (idx & 3) * 8;
        *(v4u*)&As[r * 32 + c8] = a_reg[i];
      }
    }
#endif
    {  // B tile 32x64 [k][n] -> Bs[n][k]: 2 x b128 loads, then scatter stores
      v4u b_reg[2];
      for (int i = 0; i < 2; i++) {
        int idx = tid + i * 128;
        int r = idx >> 3, c8 = (idx & 7) * 8;
        b_reg[i] = *(const v4u*)(Xb + (size_t)(k0 + r) * 1024 + bn + c8);
      }
      for (int i = 0; i < 2; i++) {
        int idx = tid + i * 128;
        int r = idx >> 3, c8 = (idx & 7) * 8;
        for (int j = 0; j < 4; j++) {
          u32 w = b_reg[i][j];
          Bs[(c8 + 2 * j + 0) * 32 + r] = (u16)w;
          Bs[(c8 + 2 * j + 1) * 32 + r] = (u16)(w >> 16);
        }
      }
    }
    if (k0 + 32 < K) {  // next tiles toward L2 (global_prefetch_b8)
      __builtin_prefetch(W + (size_t)(bm + (tid >> 1)) * K + k0 + 32, 0, 1);
      __builtin_prefetch(Xb + (size_t)(k0 + 32 + (tid >> 2)) * 1024 + bn, 0, 1);
    }
#if USE_TDM
    if (wave == 0) __builtin_amdgcn_s_wait_tensorcnt(0);
#endif
    __syncthreads();

    v16bf a = frag_split(&As[(wave * 16 + l16) * 32 + hi * 8]);
    for (int nt = 0; nt < 4; nt++) {
      v16bf b = frag_contig(&Bs[(nt * 16 + l16) * 32 + hi * 8]);
      acc[nt] = wmma_bf16(a, b, acc[nt]);
    }
  }

  for (int nt = 0; nt < 4; nt++)
    for (int i = 0; i < 8; i++) {
      int m = bm + wave * 16 + i + hi * 8;   // C layout: lanes>=16 = rows +8
      int n = bn + nt * 16 + l16;
      float vv = acc[nt][i] + bias[m];
      if (mode == 0)
        ((u16*)out)[(size_t)bat * M * 1024 + (size_t)m * 1024 + n] = f2bf(vv);
      else if (mode == 1)
        ((u16*)out)[((size_t)bat * 1024 + n) * M + m] = f2bf(vv * scale);
      else
        ((float*)out)[(size_t)bat * M * 1024 + (size_t)m * 1024 + n] = vv;
    }
}

// ---------------------------------------------------------------------------
// Flash attention. Block = (b, h, 64-row q tile), 4 waves, 16 k-tiles of 64.
// Qt: [b][n][c] (pre-scaled 1/8), Kc: [b][c][n], Vt: [b][n][c].
// Of: [b][h][q][d] flat bf16 (== [b][512][1024] for the final projection).
// ---------------------------------------------------------------------------
__global__ __launch_bounds__(128) void attn_wmma(
    const u16* __restrict__ Qt, const u16* __restrict__ Kc,
    const u16* __restrict__ Vt, u16* __restrict__ Of) {
  __shared__ __align__(16) u16 Qs[64 * 64];     // [q][d]
  __shared__ __align__(16) u16 Ks[64 * 64];     // [n][d]  (B frag for S)
  __shared__ __align__(16) u16 Vs[64 * 64];     // [d][n]  (B frag for O)
  __shared__ __align__(16) u16 Ps[4][16 * 64];  // per-wave P, [q][n] (A frag)
  const int b = blockIdx.x >> 3, h = blockIdx.x & 7;
  const int qb = blockIdx.y * 64;
  const int tid = threadIdx.x, lane = tid & 31, wave = tid >> 5;
  const int l16 = lane & 15, hi = (lane >> 4) & 1;

  const u16* Qg = Qt + ((size_t)(b * 1024 + qb) * 512 + h * 64);
#if USE_TDM
  if (wave == 0) {                       // Q tile via Tensor Data Mover
    tdm_load_2d((u32)(size_t)Qs, Qg, 64, 64, 512);
    __builtin_amdgcn_s_wait_tensorcnt(0);
  }
#else
  {
    v4u q_reg[4];
    for (int i = 0; i < 4; i++) {
      int idx = tid + i * 128;
      int r = idx >> 3, c8 = (idx & 7) * 8;
      q_reg[i] = *(const v4u*)(Qg + (size_t)r * 512 + c8);
    }
    for (int i = 0; i < 4; i++) {
      int idx = tid + i * 128;
      int r = idx >> 3, c8 = (idx & 7) * 8;
      *(v4u*)&Qs[r * 64 + c8] = q_reg[i];
    }
  }
#endif

  float mrow[8], lrow[8];
  v8f acc[4] = {};
  for (int i = 0; i < 8; i++) { mrow[i] = -3.0e38f; lrow[i] = 0.f; }

  for (int kt = 0; kt < 16; kt++) {
    const int kb = kt * 64;
    __syncthreads();
    const u16* Kg = Kc + ((size_t)(b * 512 + h * 64) * 1024 + kb);
    const u16* Vg = Vt + ((size_t)(b * 1024 + kb) * 512 + h * 64);
    // Phase 1: wide global loads into registers (batched b128s)
    v4u kreg[4], vreg[4];
    for (int i = 0; i < 4; i++) {
      int idx = tid + i * 128;
      int r = idx >> 3, c8 = (idx & 7) * 8;
      kreg[i] = *(const v4u*)(Kg + (size_t)r * 1024 + c8);  // (d=r, n=c8..)
      vreg[i] = *(const v4u*)(Vg + (size_t)r * 512 + c8);   // (n=r, d=c8..)
    }
    // Phase 2: transpose-scatter into LDS (Ks[n][d], Vs[d][n])
    for (int i = 0; i < 4; i++) {
      int idx = tid + i * 128;
      int r = idx >> 3, c8 = (idx & 7) * 8;
      for (int j = 0; j < 4; j++) {
        u32 kw = kreg[i][j], vw = vreg[i][j];
        Ks[(c8 + 2 * j + 0) * 64 + r] = (u16)kw;
        Ks[(c8 + 2 * j + 1) * 64 + r] = (u16)(kw >> 16);
        Vs[(c8 + 2 * j + 0) * 64 + r] = (u16)vw;
        Vs[(c8 + 2 * j + 1) * 64 + r] = (u16)(vw >> 16);
      }
    }
    __syncthreads();

    // S(16q x 64n per wave) = Q_tile K_tile, K-dim 64 = 2 WMMA steps
    v8f s[4] = {};
    for (int k0 = 0; k0 < 64; k0 += 32) {
      v16bf a = frag_split(&Qs[(wave * 16 + l16) * 64 + k0 + hi * 8]);
      for (int nt = 0; nt < 4; nt++) {
        v16bf kf = frag_contig(&Ks[(nt * 16 + l16) * 64 + k0 + hi * 8]);
        s[nt] = wmma_bf16(a, kf, s[nt]);
      }
    }

    // Online softmax: rows live in (VGPR i, lane-half), xor masks 1/2/4/8
    // reduce across the correct 16-lane group on wave32.
    for (int i = 0; i < 8; i++) {
      float t = fmaxf(fmaxf(s[0][i], s[1][i]), fmaxf(s[2][i], s[3][i]));
      for (int off = 1; off < 16; off <<= 1)
        t = fmaxf(t, __shfl_xor(t, off, 32));
      float mnew = fmaxf(mrow[i], t);
      float sc = __expf(mrow[i] - mnew);
      mrow[i] = mnew;
      float rs = 0.f;
      int qrow = i + hi * 8;
      for (int nt = 0; nt < 4; nt++) {
        float p = __expf(s[nt][i] - mnew);
        rs += p;
        Ps[wave][qrow * 64 + nt * 16 + l16] = f2bf(p);  // C-layout -> A-layout
        acc[nt][i] = acc[nt][i] * sc;
      }
      for (int off = 1; off < 16; off <<= 1)
        rs += __shfl_xor(rs, off, 32);
      lrow[i] = lrow[i] * sc + rs;
    }

    // O(16q x 64d per wave) += P V^T (Ps wave-private; DS in-order per wave)
    for (int k0 = 0; k0 < 64; k0 += 32) {
      v16bf a = frag_split(&Ps[wave][l16 * 64 + k0 + hi * 8]);
      for (int dt = 0; dt < 4; dt++) {
        v16bf vf = frag_contig(&Vs[(dt * 16 + l16) * 64 + k0 + hi * 8]);
        acc[dt] = wmma_bf16(a, vf, acc[dt]);
      }
    }
  }

  for (int dt = 0; dt < 4; dt++)
    for (int i = 0; i < 8; i++) {
      int q = qb + wave * 16 + i + hi * 8;
      int d = dt * 16 + l16;
      float o = acc[dt][i] / lrow[i];
      Of[(((size_t)b * 8 + h) * 1024 + q) * 64 + d] = f2bf(o);
    }
}

// ---------------------------------------------------------------------------
extern "C" void kernel_launch(void* const* d_in, const int* in_sizes, int n_in,
                              void* d_out, int out_size, void* d_ws, size_t ws_size,
                              hipStream_t stream) {
  (void)in_sizes; (void)n_in; (void)out_size; (void)ws_size;
  const float* x   = (const float*)d_in[0];
  const float* y   = (const float*)d_in[1];
  const float* w_q = (const float*)d_in[2];
  const float* b_q = (const float*)d_in[3];
  const float* w_k = (const float*)d_in[4];
  const float* b_k = (const float*)d_in[5];
  const float* w_v = (const float*)d_in[6];
  const float* b_v = (const float*)d_in[7];
  const float* w_o = (const float*)d_in[8];
  const float* b_o = (const float*)d_in[9];
  float* out = (float*)d_out;

  u16* ws = (u16*)d_ws;
  size_t off = 0;
  u16* xb  = ws + off; off += 8ull * 512 * 1024;   // x  bf16
  u16* yb  = ws + off; off += 8ull * 768 * 1024;   // y  bf16
  u16* wqb = ws + off; off += 512ull * 512;
  u16* wkb = ws + off; off += 512ull * 768;
  u16* wvb = ws + off; off += 512ull * 768;
  u16* wob = ws + off; off += 512ull * 512;
  u16* Qt  = ws + off; off += 8ull * 1024 * 512;   // Q transposed, /8
  u16* Kc  = ws + off; off += 8ull * 512 * 1024;   // K row-major
  u16* Vt  = ws + off; off += 8ull * 1024 * 512;   // V transposed
  u16* Of  = ws + off; off += 8ull * 512 * 1024;   // attention output

  auto cvt = [&](const float* s, u16* d, size_t n) {
    cvt_bf16<<<dim3((unsigned)((n + 255) / 256)), dim3(256), 0, stream>>>(s, d, (int)n);
  };
  cvt(x,   xb,  8ull * 512 * 1024);
  cvt(y,   yb,  8ull * 768 * 1024);
  cvt(w_q, wqb, 512ull * 512);
  cvt(w_k, wkb, 512ull * 768);
  cvt(w_v, wvb, 512ull * 768);
  cvt(w_o, wob, 512ull * 512);

  gemm_wmma<<<dim3(8, 16, 8), dim3(128), 0, stream>>>(wqb, xb, b_q, Qt, 512, 512, 1, 0.125f);
  gemm_wmma<<<dim3(8, 16, 8), dim3(128), 0, stream>>>(wkb, yb, b_k, Kc, 512, 768, 0, 1.0f);
  gemm_wmma<<<dim3(8, 16, 8), dim3(128), 0, stream>>>(wvb, yb, b_v, Vt, 512, 768, 1, 1.0f);

  attn_wmma<<<dim3(64, 16), dim3(128), 0, stream>>>(Qt, Kc, Vt, Of);

  gemm_wmma<<<dim3(8, 16, 8), dim3(128), 0, stream>>>(wob, Of, b_o, out, 512, 512, 2, 1.0f);
}
